// SelfAttention_85091892068967
// MI455X (gfx1250) — matrix operand order repaired and checked by reading, not compiled, not run on hardware
//
#include <hip/hip_runtime.h>
#include <hip/hip_bf16.h>
#include <math.h>

// CDNA5 (gfx1250) wave32 WMMA types
typedef __attribute__((ext_vector_type(16))) __bf16 v16bf;
typedef __attribute__((ext_vector_type(8)))  float  v8f;

union Frag {
    v16bf v;
    __bf16 bf[16];
    unsigned short u[16];
    uint2 d2[4];
    uint4 q[2];
};

// f32 -> bf16 via native conversion (clang selects packed bf16 cvt on gfx1250)
__device__ __forceinline__ unsigned short bfbits(float f) {
    union { __bf16 b; unsigned short u; } c;
    c.b = (__bf16)f;
    return c.u;
}

__device__ __forceinline__ void cvt8(const float4 a, const float4 b, __bf16* d) {
    d[0] = (__bf16)a.x; d[1] = (__bf16)a.y; d[2] = (__bf16)a.z; d[3] = (__bf16)a.w;
    d[4] = (__bf16)b.x; d[5] = (__bf16)b.y; d[6] = (__bf16)b.z; d[7] = (__bf16)b.w;
}

// CDNA5 VOP3P packed bf16 max (ISA 15.10 op 44): two row-maxes per dword
__device__ __forceinline__ unsigned pk_max_bf16(unsigned a, unsigned b) {
    unsigned d;
    __asm__("v_pk_max_num_bf16 %0, %1, %2" : "=v"(d) : "v"(a), "v"(b));
    return d;
}

// ---------------------------------------------------------------------------
// GEMM: D[M,N] = A[M,K] * W[N,K]^T + bias[N]
// One wave per 32x32 output tile (2x2 WMMA register blocking, 4 accumulators).
// A_IS_F32: A operand is f32 (convert to bf16 on the fly) else bf16 (u16).
// OUT_F32 : write f32, else write bf16 (u16).
// ALSO_T  : additionally scatter a d-major transposed bf16 copy for attention V:
//           DT laid out as [32 slices][64 d][2048 keys] (EMB=1024,S=2048,hd=64).
// ---------------------------------------------------------------------------
template <bool A_IS_F32, bool OUT_F32, bool ALSO_T>
__global__ __launch_bounds__(128)
void wmma_gemm_kernel(const void* __restrict__ Aptr,
                      const float* __restrict__ W,
                      const float* __restrict__ bias,
                      void* __restrict__ Dptr,
                      unsigned short* __restrict__ DT,
                      int N, int K, int tilesN)
{
    const int lane = threadIdx.x & 31;
    const int wid  = threadIdx.x >> 5;
    const int tile = blockIdx.x * 4 + wid;
    const int tn = tile % tilesN;
    const int tm = tile / tilesN;
    const int row0 = tm * 32, col0 = tn * 32;
    const int hl = lane >> 4;   // half-wave id (fragment layout split)
    const int lr = lane & 15;

    const float* Af          = (const float*)Aptr;
    const unsigned short* Ab = (const unsigned short*)Aptr;

    const v8f vzero = {0.f, 0.f, 0.f, 0.f, 0.f, 0.f, 0.f, 0.f};
    v8f acc00 = vzero, acc01 = vzero, acc10 = vzero, acc11 = vzero;

    const int ar0 = row0 + lr, ar1 = row0 + 16 + lr;   // A rows (A-frag: M = lane&15)
    const int wr0 = col0 + lr, wr1 = col0 + 16 + lr;   // weight rows = out columns

    for (int k0 = 0; k0 < K; k0 += 32) {
        Frag a0, a1, b0, b1;
        // A frags, 16-bit A layout: bf[0..7] <-> k0+hl*8+i ; bf[8..15] <-> k0+16+hl*8+i
        if (A_IS_F32) {
            const float4* p0 = (const float4*)(Af + (size_t)ar0 * K + k0 + hl * 8);
            cvt8(p0[0], p0[1], &a0.bf[0]);
            cvt8(p0[4], p0[5], &a0.bf[8]);
            const float4* p1 = (const float4*)(Af + (size_t)ar1 * K + k0 + hl * 8);
            cvt8(p1[0], p1[1], &a1.bf[0]);
            cvt8(p1[4], p1[5], &a1.bf[8]);
        } else {
            const unsigned short* p0 = Ab + (size_t)ar0 * K + k0 + hl * 8;
            a0.q[0] = *(const uint4*)(p0);
            a0.q[1] = *(const uint4*)(p0 + 16);
            const unsigned short* p1 = Ab + (size_t)ar1 * K + k0 + hl * 8;
            a1.q[0] = *(const uint4*)(p1);
            a1.q[1] = *(const uint4*)(p1 + 16);
        }
        // B frags (= W^T), 16-bit B layout: bf[0..7] <-> k0+hl*8+i ; bf[8..15] <-> k0+16+hl*8+i, col = lr
        {
            const float4* p = (const float4*)(W + (size_t)wr0 * K + k0 + hl * 8);
            cvt8(p[0], p[1], &b0.bf[0]);
            cvt8(p[4], p[5], &b0.bf[8]);
        }
        {
            const float4* p = (const float4*)(W + (size_t)wr1 * K + k0 + hl * 8);
            cvt8(p[0], p[1], &b1.bf[0]);
            cvt8(p[4], p[5], &b1.bf[8]);
        }
        acc00 = __builtin_amdgcn_wmma_f32_16x16x32_bf16(false, a0.v, false, b0.v,
                                                        (short)0, acc00, false, false);
        acc01 = __builtin_amdgcn_wmma_f32_16x16x32_bf16(false, a0.v, false, b1.v,
                                                        (short)0, acc01, false, false);
        acc10 = __builtin_amdgcn_wmma_f32_16x16x32_bf16(false, a1.v, false, b0.v,
                                                        (short)0, acc10, false, false);
        acc11 = __builtin_amdgcn_wmma_f32_16x16x32_bf16(false, a1.v, false, b1.v,
                                                        (short)0, acc11, false, false);
    }

    // Epilogue. D layout: VGPR r <-> row (r + 8*hl), col = lane&15 (per quadrant)
    const float bv0 = bias[col0 + lr];
    const float bv1 = bias[col0 + 16 + lr];

    auto emit = [&](const v8f& acc, int rbase, int cbase, float bv) {
#pragma unroll
        for (int r = 0; r < 8; ++r) {
            const float val = acc[r] + bv;
            const int row = rbase + r + 8 * hl;
            const int col = cbase;
            if (OUT_F32) {
                ((float*)Dptr)[(size_t)row * N + col] = val;
            } else {
                ((unsigned short*)Dptr)[(size_t)row * N + col] = bfbits(val);
            }
            if (ALSO_T) {
                // flat reinterpretation [4096,1024] -> [32][2048][64]; transpose to
                // [32][64][2048] so attention V fragments load contiguously.
                const int flat  = row * 1024 + col;
                const int slice = flat >> 17;          // / (2048*64)
                const int rem   = flat & 131071;
                const int key   = rem >> 6;
                const int d     = rem & 63;
                DT[((size_t)slice << 17) + (d << 11) + key] = bfbits(val);
            }
        }
    };
    emit(acc00, row0,      col0,      bv0);
    emit(acc01, row0,      col0 + 16, bv1);
    emit(acc10, row0 + 16, col0,      bv0);
    emit(acc11, row0 + 16, col0 + 16, bv1);
}

// ---------------------------------------------------------------------------
// Flash attention over one (head,batch) slice of the flat-reinterpreted q
// buffer: qh = qb viewed as [32][2048][64] bf16, Q=K=V=qh (faithful bug).
// qbT is the d-major copy [32][64][2048] used for vectorized V fragments.
// One wave per 16-query tile; online softmax over j-blocks of 32 keys.
// Row sums via WMMA against ones; row max via packed-bf16 butterfly.
// ---------------------------------------------------------------------------
__global__ __launch_bounds__(128)
void flash_attn_kernel(const unsigned short* __restrict__ qb,
                       const unsigned short* __restrict__ qbT,
                       unsigned short* __restrict__ out)
{
    __shared__ unsigned short Pl[4][16 * 40];   // per-wave P tile, padded stride 40

    const int lane = threadIdx.x & 31;
    const int wid  = threadIdx.x >> 5;
    const int w    = blockIdx.x * 4 + wid;      // 0..4095
    const int hb   = w >> 7;                    // (h*B+b) slice: 0..31
    const int qt   = w & 127;                   // query tile in slice
    const int hl   = lane >> 4;
    const int lr   = lane & 15;
    const int i0   = qt * 16;

    const unsigned short* base  = qb  + (size_t)hb * 2048 * 64;
    const unsigned short* baseT = qbT + (size_t)hb * 2048 * 64;
    unsigned short* myP = &Pl[wid][0];

    // Q fragments (A layout), two K-chunks covering d = 0..63
    Frag aQ[2];
#pragma unroll
    for (int c = 0; c < 2; ++c) {
        const unsigned short* p = base + (size_t)(i0 + lr) * 64 + c * 32 + hl * 8;
        aQ[c].q[0] = *(const uint4*)(p);
        aQ[c].q[1] = *(const uint4*)(p + 16);
    }

    // Constant all-ones B fragment: rowsum(P) = P * ones via one WMMA
    Frag bOne;
#pragma unroll
    for (int i = 0; i < 16; ++i) bOne.u[i] = 0x3F80;   // bf16 1.0

    float mi[8], li[8];
    v8f O[4];
    const v8f vzero = {0.f, 0.f, 0.f, 0.f, 0.f, 0.f, 0.f, 0.f};
#pragma unroll
    for (int c = 0; c < 4; ++c) O[c] = vzero;
#pragma unroll
    for (int r = 0; r < 8; ++r) { mi[r] = -INFINITY; li[r] = 0.f; }

    const float scale = 0.03125f;   // 1/sqrt(EMB) = 1/32 (faithful: sqrt(emb), not hd)

    for (int jb = 0; jb < 2048; jb += 32) {
        // ---- prefetch next j-block's K rows and d-major V slice into near cache
        if (jb + 32 < 2048) {
            __builtin_prefetch(base  + (size_t)(jb + 32 + lane) * 64, 0, 3);
            __builtin_prefetch(baseT + (size_t)lane * 2048 + jb + 32, 0, 3);
            __builtin_prefetch(baseT + (size_t)(32 + lane) * 2048 + jb + 32, 0, 3);
        }

        // ---- load ALL four K fragments first (distinct regs -> pipelined loads)
        Frag bk[2][2];   // [jhalf][kchunk]
#pragma unroll
        for (int c = 0; c < 2; ++c) {
            const unsigned short* p0 = base + (size_t)(jb + lr) * 64 + c * 32 + hl * 8;
            const unsigned short* p1 = base + (size_t)(jb + 16 + lr) * 64 + c * 32 + hl * 8;
            bk[0][c].q[0] = *(const uint4*)(p0);  bk[0][c].q[1] = *(const uint4*)(p0 + 16);
            bk[1][c].q[0] = *(const uint4*)(p1);  bk[1][c].q[1] = *(const uint4*)(p1 + 16);
        }
        // ---- scores: S0 = Q·K[jb..jb+15]^T, S1 = Q·K[jb+16..jb+31]^T
        v8f s0, s1;
        s0 = __builtin_amdgcn_wmma_f32_16x16x32_bf16(false, aQ[0].v, false, bk[0][0].v,
                                                     (short)0, vzero, false, false);
        s1 = __builtin_amdgcn_wmma_f32_16x16x32_bf16(false, aQ[0].v, false, bk[1][0].v,
                                                     (short)0, vzero, false, false);
        s0 = __builtin_amdgcn_wmma_f32_16x16x32_bf16(false, aQ[1].v, false, bk[0][1].v,
                                                     (short)0, s0, false, false);
        s1 = __builtin_amdgcn_wmma_f32_16x16x32_bf16(false, aQ[1].v, false, bk[1][1].v,
                                                     (short)0, s1, false, false);

        // ---- row max: packed-bf16 butterfly over the 16 lanes of this half-wave
        float v0[8], v1[8];
#pragma unroll
        for (int r = 0; r < 8; ++r) { v0[r] = s0[r] * scale; v1[r] = s1[r] * scale; }
        unsigned pk[4];
#pragma unroll
        for (int r2 = 0; r2 < 4; ++r2) {
            const float mA = fmaxf(v0[2 * r2],     v1[2 * r2]);
            const float mB = fmaxf(v0[2 * r2 + 1], v1[2 * r2 + 1]);
            pk[r2] = (unsigned)bfbits(mA) | ((unsigned)bfbits(mB) << 16);
        }
#pragma unroll
        for (int mask = 1; mask <= 8; mask <<= 1) {
#pragma unroll
            for (int r2 = 0; r2 < 4; ++r2)
                pk[r2] = pk_max_bf16(pk[r2], (unsigned)__shfl_xor((int)pk[r2], mask, 32));
        }
        float corr[8], p0v[8], p1v[8];
#pragma unroll
        for (int r = 0; r < 8; ++r) {
            const unsigned bits = (r & 1) ? (pk[r >> 1] & 0xFFFF0000u)
                                          : (pk[r >> 1] << 16);
            const float rowmax = __uint_as_float(bits);
            const float newm = fmaxf(mi[r], rowmax);
            corr[r] = __expf(mi[r] - newm);
            p0v[r]  = __expf(v0[r] - newm);
            p1v[r]  = __expf(v1[r] - newm);
            mi[r] = newm;
        }

        // ---- transpose P (D layout -> A layout) through per-wave LDS tile
#pragma unroll
        for (int r = 0; r < 8; ++r) {
            const int row = r + 8 * hl;
            myP[row * 40 + lr]      = bfbits(p0v[r]);
            myP[row * 40 + 16 + lr] = bfbits(p1v[r]);
        }
        __asm__ volatile("s_wait_dscnt 0" ::: "memory");  // LDS in-order per wave
        Frag aP;
        {
            const unsigned short* p = myP + lr * 40 + hl * 8;   // 8B-aligned
            aP.d2[0] = *(const uint2*)(p);
            aP.d2[1] = *(const uint2*)(p + 4);
            aP.d2[2] = *(const uint2*)(p + 16);
            aP.d2[3] = *(const uint2*)(p + 20);
        }

        // ---- load ALL four V fragments (distinct regs), then rowsum + O updates
        Frag bV[4];
#pragma unroll
        for (int c = 0; c < 4; ++c) {
            const unsigned short* p = baseT + (size_t)(c * 16 + lr) * 2048 + jb + hl * 8;
            bV[c].q[0] = *(const uint4*)(p);
            bV[c].q[1] = *(const uint4*)(p + 16);
        }

        // row sums of (bf16-rounded) P via one WMMA: rs[r] = sum_j P[row][j]
        v8f rs = __builtin_amdgcn_wmma_f32_16x16x32_bf16(false, aP.v, false, bOne.v,
                                                         (short)0, vzero, false, false);
#pragma unroll
        for (int r = 0; r < 8; ++r) li[r] = li[r] * corr[r] + rs[r];

        // rescale O then accumulate O += P·V
#pragma unroll
        for (int c = 0; c < 4; ++c) {
#pragma unroll
            for (int r = 0; r < 8; ++r) O[c][r] *= corr[r];
        }
#pragma unroll
        for (int c = 0; c < 4; ++c)
            O[c] = __builtin_amdgcn_wmma_f32_16x16x32_bf16(false, aP.v, false, bV[c].v,
                                                           (short)0, O[c], false, false);
    }

    // ---- normalize and store (flat layout feeds GEMM2 directly = faithful reshape)
#pragma unroll
    for (int c = 0; c < 4; ++c) {
#pragma unroll
        for (int r = 0; r < 8; ++r) {
            const float o = O[c][r] / li[r];
            out[(size_t)hb * 2048 * 64 + (size_t)(i0 + r + 8 * hl) * 64 + c * 16 + lr] =
                bfbits(o);
        }
    }
}

// ---------------------------------------------------------------------------
extern "C" void kernel_launch(void* const* d_in, const int* in_sizes, int n_in,
                              void* d_out, int out_size, void* d_ws, size_t ws_size,
                              hipStream_t stream)
{
    const float* x  = (const float*)d_in[0];  // [2,2048,1024]
    const float* wq = (const float*)d_in[1];  // [1024,1024] (out,in)
    const float* bq = (const float*)d_in[2];  // [1024]
    const float* wu = (const float*)d_in[3];  // [1024,1024]
    const float* bu = (const float*)d_in[4];  // [1024]

    unsigned short* qb  = (unsigned short*)d_ws;                  // 8 MB
    unsigned short* qbT = qb  + (size_t)4096 * 1024;              // 8 MB (d-major copy)
    unsigned short* att = qbT + (size_t)4096 * 1024;              // 8 MB

    const dim3 blk(128);            // 4 wave32 waves / block
    const int  tilesN = 1024 / 32;  // 32 (32-wide output tiles)
    const int  nTiles = (4096 / 32) * tilesN;   // 4096 tiles of 32x32
    const int  gemmBlocks = nTiles / 4;         // 1024

    // q = x * wq^T + bq  (f32 in, bf16 out, plus d-major transposed copy)
    wmma_gemm_kernel<true, false, true><<<gemmBlocks, blk, 0, stream>>>(
        (const void*)x, wq, bq, (void*)qb, qbT, 1024, 1024, tilesN);

    // attention on flat-reinterpreted qh (Q=K=V), flash-style softmax
    flash_attn_kernel<<<1024, blk, 0, stream>>>(qb, qbT, att);

    // y = att * wu^T + bu  (bf16 in, f32 out)
    wmma_gemm_kernel<false, true, false><<<gemmBlocks, blk, 0, stream>>>(
        (const void*)att, wu, bu, d_out, nullptr, 1024, 1024, tilesN);
}
